// MambaBlock_47321949667956
// MI455X (gfx1250) — matrix operand order repaired and checked by reading, not compiled, not run on hardware
//
#include <hip/hip_runtime.h>
#include <hip/hip_bf16.h>
#include <math.h>

// ---------------------------------------------------------------------------
// Mamba block (2 layers) for MI455X / gfx1250.
//   * All four GEMMs via v_wmma_f32_16x16x32_bf16, 32x64 tile per wave,
//     pure-bf16 b128 loads in the inner loop (operands pre-converted once;
//     conversions fused into producer epilogues where possible).
//   * Selective scan parallelized as a 3-pass chunked linear scan
//     (64 chunks x 64 steps) to avoid a ~100us serial-latency wall.
//   * Working set (~160MB) lives in d_ws, mostly resident in the 192MB L2.
// ---------------------------------------------------------------------------

#define BATCH   2
#define LSEQ    4096
#define DMODEL  512
#define DINNER  1024
#define NSTATE  16
#define DTRANK  32
#define DCONV   4
#define NLAYER  2
#define LCHUNK  64
#define NCHUNK  (LSEQ / LCHUNK)
#define ROWS    (BATCH * LSEQ)

typedef __attribute__((ext_vector_type(16))) __bf16 v16bf;
typedef __attribute__((ext_vector_type(8)))  __bf16 v8bf;
typedef __attribute__((ext_vector_type(8)))  float  v8f;

__device__ __forceinline__ float silu_f(float x) {
  return x / (1.0f + __expf(-x));
}
__device__ __forceinline__ float softplus_f(float x) {
  return (x > 20.0f) ? x : log1pf(__expf(x));
}

// --- WMMA fragment loaders (contiguous k order, same for A and B), bf16 source ------
// A: 16x32 (MxK). lane g=lane>>4, m=lane&15:
//   elems 0..15 <- K = k0 + g*16 .. +15 (one 32B load)
__device__ __forceinline__ v16bf frag_a(const __bf16* __restrict__ A, int lda,
                                        int m0, int k0, int lane) {
  const int g = lane >> 4, m = lane & 15;
  const __bf16* p = A + (size_t)(m0 + m) * lda + k0 + g * 16;
  return *(const v16bf*)p;
}
// B: 32x16 (KxN), column n=lane&15; lanes 0-15 hold K=k0..15, lanes 16-31
// hold K=k0+16..31. W is [N_out,K] row-major (B^T), so 16 contiguous bf16.
__device__ __forceinline__ v16bf frag_b(const __bf16* __restrict__ W, int ldb,
                                        int n0, int k0, int lane) {
  const int g = lane >> 4, c = lane & 15;
  const __bf16* p = W + (size_t)(n0 + c) * ldb + k0 + g * 16;
  return *(const v16bf*)p;
}

// --- GEMM: C[M,N] = A[M,K] @ W[N,K]^T, one 32x64 tile per wave --------------
// mode 0: plain f32 store.  mode 1: softplus(val + bias[col]) (dt_proj).
__global__ __launch_bounds__(256) void gemm_wmma_bf16(
    const __bf16* __restrict__ A, const __bf16* __restrict__ W,
    float* __restrict__ C, int M, int N, int K,
    int lda, int ldb, int ldc,
    const float* __restrict__ bias, int mode) {
  const int wave = threadIdx.x >> 5;
  const int lane = threadIdx.x & 31;
  const int tilesN = N >> 6;                 // 64 columns per wave
  const int t = blockIdx.x * 8 + wave;
  if (t >= (M >> 5) * tilesN) return;        // 32 rows per wave
  const int m0 = (t / tilesN) << 5;
  const int n0 = (t % tilesN) << 6;

  v8f acc0[4] = {};
  v8f acc1[4] = {};
  for (int k0 = 0; k0 < K; k0 += 32) {
    const v16bf a0 = frag_a(A, lda, m0,      k0, lane);
    const v16bf a1 = frag_a(A, lda, m0 + 16, k0, lane);
#pragma unroll
    for (int j = 0; j < 4; ++j) {
      const v16bf b = frag_b(W, ldb, n0 + j * 16, k0, lane);
      acc0[j] = __builtin_amdgcn_wmma_f32_16x16x32_bf16(
          false, a0, false, b, (short)0, acc0[j], false, false);
      acc1[j] = __builtin_amdgcn_wmma_f32_16x16x32_bf16(
          false, a1, false, b, (short)0, acc1[j], false, false);
    }
  }
  const int g = lane >> 4, c = lane & 15;
#pragma unroll
  for (int j = 0; j < 4; ++j) {
    const int col = n0 + j * 16 + c;
    const float bv = (mode == 1) ? bias[col] : 0.0f;
#pragma unroll
    for (int v = 0; v < 8; ++v) {
      const int r0 = m0 + g * 8 + v;
      float v0 = acc0[j][v], v1 = acc1[j][v];
      if (mode == 1) { v0 = softplus_f(v0 + bv); v1 = softplus_f(v1 + bv); }
      C[(size_t)r0 * ldc + col]        = v0;
      C[(size_t)(r0 + 16) * ldc + col] = v1;
    }
  }
}

// --- elementwise f32 -> bf16 (x4 per thread) --------------------------------
__global__ __launch_bounds__(256) void f32_to_bf16(
    const float* __restrict__ inp, __bf16* __restrict__ outp) {
  const size_t i = ((size_t)blockIdx.x * 256 + threadIdx.x) * 4;
  const float4 v = *(const float4*)(inp + i);
  v8bf* o2 = (v8bf*)nullptr; (void)o2;
  __bf16* o = outp + i;
  o[0] = (__bf16)v.x; o[1] = (__bf16)v.y; o[2] = (__bf16)v.z; o[3] = (__bf16)v.w;
}

// --- depthwise causal conv (width 4) + SiLU; writes f32 + bf16 shadow -------
__global__ __launch_bounds__(256) void dwconv_silu(
    const float* __restrict__ xz, const float* __restrict__ cw,
    const float* __restrict__ cb, float* __restrict__ xc,
    __bf16* __restrict__ xcb) {
  const int idx = blockIdx.x * 256 + threadIdx.x;   // (b*L+l)*DINNER + d
  const int d  = idx & (DINNER - 1);
  const int rl = idx >> 10;                         // b*L + l
  const int l  = rl & (LSEQ - 1);
  const float* w = cw + d * DCONV;
  float acc = cb[d];
#pragma unroll
  for (int j = 0; j < DCONV; ++j) {
    const int ls = l + j - (DCONV - 1);
    if (ls >= 0)
      acc += w[j] * xz[(size_t)(rl + j - (DCONV - 1)) * (2 * DINNER) + d];
  }
  const float y = silu_f(acc);
  xc[(size_t)rl * DINNER + d]  = y;
  xcb[(size_t)rl * DINNER + d] = (__bf16)y;
}

// --- Selective scan, 3-pass chunked linear scan -----------------------------
__global__ __launch_bounds__(256) void scan_pass1(
    const float* __restrict__ dt,    // xz base, ld = 2*DINNER (cols 0..DINNER)
    const float* __restrict__ xc,
    const float* __restrict__ xdbl,  // [ROWS,64]: [32:48)=B, [48:64)=C
    const float* __restrict__ A_log,
    float* __restrict__ prodA, float* __restrict__ hloc) {
  __shared__ float sBC[2 * NSTATE];
  const int tid  = threadIdx.x;
  const int gidx = blockIdx.x * 256 + tid;   // b*NCHUNK*DINNER + ch*DINNER + d
  const int d  = gidx & (DINNER - 1);
  const int ch = (gidx >> 10) & (NCHUNK - 1);
  const int b  = gidx >> 16;

  float Arow[NSTATE], h[NSTATE], pr[NSTATE];
#pragma unroll
  for (int n = 0; n < NSTATE; ++n) {
    Arow[n] = -__expf(A_log[d * NSTATE + n]);
    h[n] = 0.0f; pr[n] = 1.0f;
  }
  for (int i = 0; i < LCHUNK; ++i) {
    const int row = b * LSEQ + ch * LCHUNK + i;
    if (tid < 2 * NSTATE) sBC[tid] = xdbl[(size_t)row * 64 + DTRANK + tid];
    __syncthreads();
    if (tid < 2 * NSTATE && i + 1 < LCHUNK)
      __builtin_prefetch(&xdbl[(size_t)(row + 1) * 64 + DTRANK + tid], 0, 0);
    const float dtv = dt[(size_t)row * (2 * DINNER) + d];
    const float dux = dtv * xc[(size_t)row * DINNER + d];
#pragma unroll
    for (int n = 0; n < NSTATE; ++n) {
      const float dA = __expf(dtv * Arow[n]);
      pr[n] *= dA;
      h[n] = dA * h[n] + dux * sBC[n];
    }
    __syncthreads();
  }
  const size_t o = (size_t)gidx * NSTATE;
#pragma unroll
  for (int n = 0; n < NSTATE; ++n) { prodA[o + n] = pr[n]; hloc[o + n] = h[n]; }
}

__global__ __launch_bounds__(256) void scan_pass2(
    float* __restrict__ prodA, const float* __restrict__ hloc) {
  const int gidx = blockIdx.x * 256 + threadIdx.x;  // b*DINNER*NSTATE + d*16 + n
  const int n = gidx & (NSTATE - 1);
  const int d = (gidx >> 4) & (DINNER - 1);
  const int b = gidx >> 14;
  float h = 0.0f;
  for (int ch = 0; ch < NCHUNK; ++ch) {
    const size_t o = ((size_t)((b * NCHUNK + ch) * DINNER + d)) * NSTATE + n;
    const float p  = prodA[o];
    const float hl = hloc[o];
    prodA[o] = h;            // becomes initial state H0 for this chunk
    h = p * h + hl;
  }
}

// Pass 3: replay chunk from H0; fuse y = scan + D*x, y *= silu(z).
// Writes y over xc (f32) and the bf16 shadow for the out_proj GEMM.
__global__ __launch_bounds__(256) void scan_pass3(
    const float* __restrict__ dt, const float* __restrict__ xz,  // z at +DINNER
    float* __restrict__ xc, __bf16* __restrict__ xcb,
    const float* __restrict__ xdbl,
    const float* __restrict__ A_log, const float* __restrict__ Dp,
    const float* __restrict__ h0buf) {
  __shared__ float sBC[2 * NSTATE];
  const int tid  = threadIdx.x;
  const int gidx = blockIdx.x * 256 + tid;
  const int d  = gidx & (DINNER - 1);
  const int ch = (gidx >> 10) & (NCHUNK - 1);
  const int b  = gidx >> 16;

  float Arow[NSTATE], h[NSTATE];
  const size_t o = (size_t)gidx * NSTATE;
#pragma unroll
  for (int n = 0; n < NSTATE; ++n) {
    Arow[n] = -__expf(A_log[d * NSTATE + n]);
    h[n] = h0buf[o + n];
  }
  const float Dv = Dp[d];
  for (int i = 0; i < LCHUNK; ++i) {
    const int row = b * LSEQ + ch * LCHUNK + i;
    if (tid < 2 * NSTATE) sBC[tid] = xdbl[(size_t)row * 64 + DTRANK + tid];
    __syncthreads();
    if (tid < 2 * NSTATE && i + 1 < LCHUNK)
      __builtin_prefetch(&xdbl[(size_t)(row + 1) * 64 + DTRANK + tid], 0, 0);
    const float dtv = dt[(size_t)row * (2 * DINNER) + d];
    const float xv  = xc[(size_t)row * DINNER + d];
    const float dux = dtv * xv;
    float y = 0.0f;
#pragma unroll
    for (int n = 0; n < NSTATE; ++n) {
      const float dA = __expf(dtv * Arow[n]);
      h[n] = dA * h[n] + dux * sBC[n];
      y += h[n] * sBC[NSTATE + n];
    }
    y += Dv * xv;
    const float z = xz[(size_t)row * (2 * DINNER) + DINNER + d];
    y *= silu_f(z);
    __syncthreads();
    xc[(size_t)row * DINNER + d]  = y;
    xcb[(size_t)row * DINNER + d] = (__bf16)y;
  }
}

// --- LayerNorm over 512 cols; optional bf16 shadow for next layer's GEMM ----
__global__ __launch_bounds__(256) void layer_norm(
    const float* __restrict__ inp, int ldi,
    const float* __restrict__ gma, const float* __restrict__ bta,
    float* __restrict__ outp, __bf16* __restrict__ obf) {
  __shared__ float s1[256], s2[256];
  const int row = blockIdx.x;
  const int tid = threadIdx.x;
  const float a = inp[(size_t)row * ldi + tid];
  const float b = inp[(size_t)row * ldi + tid + 256];
  s1[tid] = a + b;
  s2[tid] = a * a + b * b;
  __syncthreads();
  for (int st = 128; st > 0; st >>= 1) {
    if (tid < st) { s1[tid] += s1[tid + st]; s2[tid] += s2[tid + st]; }
    __syncthreads();
  }
  const float mu   = s1[0] * (1.0f / DMODEL);
  const float var  = s2[0] * (1.0f / DMODEL) - mu * mu;
  const float rstd = rsqrtf(var + 1e-5f);
  const float o0 = (a - mu) * rstd * gma[tid]       + bta[tid];
  const float o1 = (b - mu) * rstd * gma[tid + 256] + bta[tid + 256];
  outp[(size_t)row * DMODEL + tid]       = o0;
  outp[(size_t)row * DMODEL + tid + 256] = o1;
  if (obf) {
    obf[(size_t)row * DMODEL + tid]       = (__bf16)o0;
    obf[(size_t)row * DMODEL + tid + 256] = (__bf16)o1;
  }
}

extern "C" void kernel_launch(void* const* d_in, const int* in_sizes, int n_in,
                              void* d_out, int out_size, void* d_ws, size_t ws_size,
                              hipStream_t stream) {
  const float* x      = (const float*)d_in[0];
  const float* inW    = (const float*)d_in[1];
  const float* convW  = (const float*)d_in[2];
  const float* convB  = (const float*)d_in[3];
  const float* xprojW = (const float*)d_in[4];
  const float* dtW    = (const float*)d_in[5];
  const float* dtB    = (const float*)d_in[6];
  const float* Alog   = (const float*)d_in[7];
  const float* Dp     = (const float*)d_in[8];
  const float* outW   = (const float*)d_in[9];
  const float* lnG    = (const float*)d_in[10];
  const float* lnB    = (const float*)d_in[11];
  float* out = (float*)d_out;

  // f32 workspace
  float* xz    = (float*)d_ws;                          // [ROWS, 2048]
  float* xc    = xz    + (size_t)ROWS * 2 * DINNER;     // [ROWS, 1024]
  float* xdbl  = xc    + (size_t)ROWS * DINNER;         // [ROWS, 64]
  float* xbuf  = xdbl  + (size_t)ROWS * 64;             // [ROWS, 512]
  float* prodA = xbuf  + (size_t)ROWS * DMODEL;         // [B,NCH,DI,16]
  float* hloc  = prodA + (size_t)BATCH * NCHUNK * DINNER * NSTATE;
  // bf16 workspace
  __bf16* bfBase = (__bf16*)(hloc + (size_t)BATCH * NCHUNK * DINNER * NSTATE);
  __bf16* xin_bf   = bfBase;                                    // [ROWS,512]
  __bf16* xc_bf    = xin_bf   + (size_t)ROWS * DMODEL;          // [ROWS,1024]
  __bf16* xdbl_bf  = xc_bf    + (size_t)ROWS * DINNER;          // [ROWS,64]
  __bf16* inW_bf   = xdbl_bf  + (size_t)ROWS * 64;
  __bf16* xprojW_bf= inW_bf   + (size_t)NLAYER * 2 * DINNER * DMODEL;
  __bf16* dtW_bf   = xprojW_bf+ (size_t)NLAYER * 64 * DINNER;
  __bf16* outW_bf  = dtW_bf   + (size_t)NLAYER * DINNER * DTRANK;

  // One-time (per launch) weight conversions, all layers.
  f32_to_bf16<<<(NLAYER * 2 * DINNER * DMODEL) / 1024, 256, 0, stream>>>(inW, inW_bf);
  f32_to_bf16<<<(NLAYER * 64 * DINNER) / 1024, 256, 0, stream>>>(xprojW, xprojW_bf);
  f32_to_bf16<<<(NLAYER * DINNER * DTRANK) / 1024, 256, 0, stream>>>(dtW, dtW_bf);
  f32_to_bf16<<<(NLAYER * DMODEL * DINNER) / 1024, 256, 0, stream>>>(outW, outW_bf);
  // Layer-0 input activation conversion.
  f32_to_bf16<<<((size_t)ROWS * DMODEL) / 1024, 256, 0, stream>>>(x, xin_bf);

  for (int l = 0; l < NLAYER; ++l) {
    float* lnout = (l == NLAYER - 1) ? out : xbuf;

    { // in_proj: [8192,512]bf @ [2048,512]bf^T -> xz (f32)
      const int M = ROWS, N = 2 * DINNER, K = DMODEL;
      gemm_wmma_bf16<<<(M / 32) * (N / 64) / 8, 256, 0, stream>>>(
          xin_bf, inW_bf + (size_t)l * N * K, xz, M, N, K, K, K, N, nullptr, 0);
    }
    dwconv_silu<<<(size_t)ROWS * DINNER / 256, 256, 0, stream>>>(
        xz, convW + l * DINNER * DCONV, convB + l * DINNER, xc, xc_bf);
    { // x_proj: [8192,1024]bf @ [64,1024]bf^T -> xdbl (f32)
      const int M = ROWS, N = 64, K = DINNER;
      gemm_wmma_bf16<<<(M / 32) * (N / 64) / 8, 256, 0, stream>>>(
          xc_bf, xprojW_bf + (size_t)l * N * K, xdbl, M, N, K, K, K, N, nullptr, 0);
    }
    f32_to_bf16<<<((size_t)ROWS * 64) / 1024, 256, 0, stream>>>(xdbl, xdbl_bf);
    { // dt_proj + softplus: [8192,32]bf @ [1024,32]bf^T -> xz cols [0,1024)
      const int M = ROWS, N = DINNER, K = DTRANK;
      gemm_wmma_bf16<<<(M / 32) * (N / 64) / 8, 256, 0, stream>>>(
          xdbl_bf, dtW_bf + (size_t)l * N * K, xz, M, N, K, 64, K, 2 * DINNER,
          dtB + l * DINNER, 1);
    }
    scan_pass1<<<BATCH * NCHUNK * DINNER / 256, 256, 0, stream>>>(
        xz, xc, xdbl, Alog + (size_t)l * DINNER * NSTATE, prodA, hloc);
    scan_pass2<<<BATCH * DINNER * NSTATE / 256, 256, 0, stream>>>(prodA, hloc);
    scan_pass3<<<BATCH * NCHUNK * DINNER / 256, 256, 0, stream>>>(
        xz, xz, xc, xc_bf, xdbl, Alog + (size_t)l * DINNER * NSTATE,
        Dp + l * DINNER, prodA);
    { // out_proj: [8192,1024]bf @ [512,1024]bf^T -> xz cols [0,512)
      const int M = ROWS, N = DMODEL, K = DINNER;
      gemm_wmma_bf16<<<(M / 32) * (N / 64) / 8, 256, 0, stream>>>(
          xc_bf, outW_bf + (size_t)l * N * K, xz, M, N, K, K, K, 2 * DINNER,
          nullptr, 0);
    }
    // LayerNorm -> next layer input (f32 + bf16 shadow) or final output.
    layer_norm<<<ROWS, 256, 0, stream>>>(
        xz, 2 * DINNER, lnG + l * DMODEL, lnB + l * DMODEL, lnout,
        (l == NLAYER - 1) ? (__bf16*)nullptr : xin_bf);
  }
}
